// MultiHeadAttention_45724221833477
// MI455X (gfx1250) — matrix operand order repaired and checked by reading, not compiled, not run on hardware
//
#include <hip/hip_runtime.h>

// ---------------------------------------------------------------------------
// MultiHeadAttention on MI455X (gfx1250): bf16 WMMA GEMMs + f32 flash softmax
// K/V tile staging uses gfx1250 async memory->LDS DMA (ASYNCcnt path).
// ---------------------------------------------------------------------------

#define BS   4
#define QLEN 2048
#define DIM  1024
#define NH   16
#define DH   64

typedef __attribute__((ext_vector_type(16))) __bf16 v16bf;
typedef __attribute__((ext_vector_type(8)))  float  v8f;

static constexpr float NEGF = -3.4028234663852886e38f;  // float32 min

static __device__ inline v8f zero8() {
  v8f z;
#pragma unroll
  for (int i = 0; i < 8; ++i) z[i] = 0.0f;
  return z;
}

// A fragment: 16x32 bf16 tile of a row-major matrix (row stride lda elements).
// Layout: lanes 0-15 -> rows 0-15, K 0-15 (elems 0-15);
// lanes 16-31 -> same rows, K 16-31.
static __device__ inline v16bf load_a_frag(const __bf16* A, int lda) {
  const int lane = threadIdx.x & 31;
  const int row  = lane & 15;
  const int kb   = (lane >> 4) << 4;  // 0 or 16
  const __bf16* p = A + row * lda + kb;
  v16bf a;
#pragma unroll
  for (int j = 0; j < 16; ++j) a[j] = p[j];
  return a;
}

// B fragment (32x16, KxN) where B[k][n] == W[n][k], W row-major (stride ldw).
// Lane = column n, elems j = K kb..kb+15 -> contiguous 32-byte read of W row n.
static __device__ inline v16bf load_bT_frag(const __bf16* Wtile, int ldw) {
  const int lane = threadIdx.x & 31;
  const int col  = lane & 15;
  const int kb   = (lane >> 4) << 4;  // 0 or 16
  const __bf16* p = Wtile + col * ldw + kb;
  v16bf b;
#pragma unroll
  for (int j = 0; j < 16; ++j) b[j] = p[j];
  return b;
}

// B fragment from genuinely row-major B[k][n] (stride ldb), e.g. the V tile.
static __device__ inline v16bf load_b_frag(const __bf16* Btile, int ldb) {
  const int lane = threadIdx.x & 31;
  const int col  = lane & 15;
  const int kb   = (lane >> 4) << 4;
  v16bf b;
#pragma unroll
  for (int j = 0; j < 16; ++j) b[j] = Btile[(kb + j) * ldb + col];
  return b;
}

static __device__ inline v8f wmma_bf16(v16bf a, v16bf b, v8f c) {
  return __builtin_amdgcn_wmma_f32_16x16x32_bf16(false, a, false, b,
                                                 (short)0, c, false, false);
}

// gfx1250 async global->LDS DMA (tracked with ASYNCcnt).  lds = low 32 bits of
// the generic LDS address (LDS aperture keeps the byte offset in ADDR[31:0]).
static __device__ inline void async_ld_b128(unsigned lds, const void* gaddr) {
  asm volatile("global_load_async_to_lds_b128 %0, %1, off"
               :
               : "v"(lds), "v"((unsigned long long)(uintptr_t)gaddr)
               : "memory");
}
static __device__ inline void wait_async0() {
  asm volatile("s_wait_asynccnt 0x0" ::: "memory");
}

// ---------------------------------------------------------------------------
// Kernel 1: f32 -> bf16 conversion (grid-stride)
// ---------------------------------------------------------------------------
__global__ void cvt_f32_bf16(const float* __restrict__ src,
                             __bf16* __restrict__ dst, int n) {
  int i = blockIdx.x * blockDim.x + threadIdx.x;
  const int stride = gridDim.x * blockDim.x;
  for (; i < n; i += stride) dst[i] = (__bf16)src[i];
}

// ---------------------------------------------------------------------------
// Kernel 2: fused QKV projection.  out = x @ W^T + b  (Q also * 1/sqrt(DH))
// Each wave computes one 16x64 tile of one of Q/K/V.  Output layout:
// QKV[mat][b*NH+h][q][dh]  (bf16) so attention reads are contiguous.
// ---------------------------------------------------------------------------
__global__ void qkv_proj_kernel(const __bf16* __restrict__ X,
                                const __bf16* __restrict__ W3,  // Wq,Wk,Wv stacked
                                const float* __restrict__ bq,
                                const float* __restrict__ bk,
                                const float* __restrict__ bv,
                                __bf16* __restrict__ QKV) {
  const int wave = threadIdx.x >> 5;
  const int lane = threadIdx.x & 31;
  const int NT = DIM / 64;          // 16 column tiles
  const int MT = BS * QLEN / 16;    // 512 row tiles
  const int tile = blockIdx.x * 8 + wave;          // 3*MT*NT total
  const int mat = tile / (MT * NT);
  const int rem = tile % (MT * NT);
  const int mt = rem / NT, nt = rem % NT;

  const __bf16* A0 = X + (size_t)mt * 16 * DIM;
  const __bf16* W  = W3 + (size_t)mat * DIM * DIM;

  v8f acc[4] = {zero8(), zero8(), zero8(), zero8()};
  for (int kk = 0; kk < DIM; kk += 32) {
    v16bf a = load_a_frag(A0 + kk, DIM);
#pragma unroll
    for (int t = 0; t < 4; ++t) {
      v16bf bf = load_bT_frag(W + (size_t)(nt * 64 + t * 16) * DIM + kk, DIM);
      acc[t] = wmma_bf16(a, bf, acc[t]);
    }
  }

  const float* bias = (mat == 0) ? bq : (mat == 1) ? bk : bv;
  const float scale = (mat == 0) ? 0.125f : 1.0f;   // 1/sqrt(DH)=1/8
  const int hi = lane >> 4, col = lane & 15;
#pragma unroll
  for (int t = 0; t < 4; ++t) {
    const int n  = nt * 64 + t * 16 + col;
    const int h  = n >> 6, dh = n & 63;
    const float bias_n = bias[n];
#pragma unroll
    for (int r = 0; r < 8; ++r) {
      const int m = mt * 16 + r + 8 * hi;          // == b*QLEN + q
      const int b = m >> 11, q = m & (QLEN - 1);
      const float v = (acc[t][r] + bias_n) * scale;
      QKV[(((size_t)mat * BS * NH + (size_t)b * NH + h) * QLEN + q) * DH + dh] =
          (__bf16)v;
    }
  }
}

// ---------------------------------------------------------------------------
// Kernel 3: flash attention.  128 threads = 4 waves; each wave owns 16 query
// rows; workgroup streams 64-key K/V tiles through LDS via async DMA;
// online softmax in f32.
// ---------------------------------------------------------------------------
struct alignas(16) AttnSmem {
  __bf16 K[64 * 64];       // key tile,   row-major [k][dh]
  __bf16 V[64 * 64];       // value tile, row-major [k][dh]
  float  S[4][16 * 64];    // per-wave raw scores
  __bf16 P[4][16 * 64];    // per-wave exp weights (bf16)
  float  mrow[4][16];      // running max
  float  lrow[4][16];      // running denom
  float  arow[4][16];      // per-block rescale factor
  int    maskb[64];        // attention_mask slice
};

__global__ void attn_kernel(const __bf16* __restrict__ Qten,
                            const __bf16* __restrict__ Kten,
                            const __bf16* __restrict__ Vten,
                            const int* __restrict__ amask,
                            __bf16* __restrict__ ctx) {
  __shared__ AttnSmem sm;

  const int w    = threadIdx.x >> 5;
  const int lane = threadIdx.x & 31;
  const int hi   = lane >> 4, col = lane & 15;

  const int bh = blockIdx.x / (QLEN / 64);   // b*NH + h
  const int qb = blockIdx.x % (QLEN / 64);
  const int b  = bh / NH, h = bh % NH;
  const int q0 = qb * 64 + w * 16;

  // Q fragments for this wave's 16 rows (two 32-wide K slices of DH=64)
  const __bf16* Qbase = Qten + ((size_t)bh * QLEN + q0) * DH;
  v16bf qa0 = load_a_frag(Qbase, DH);
  v16bf qa1 = load_a_frag(Qbase + 32, DH);

  if (lane < 16) { sm.mrow[w][lane] = NEGF; sm.lrow[w][lane] = 0.0f; }

  const unsigned ldsK = (unsigned)(uintptr_t)(void*)sm.K;
  const unsigned ldsV = (unsigned)(uintptr_t)(void*)sm.V;

  v8f acc[4] = {zero8(), zero8(), zero8(), zero8()};

  for (int kb = 0; kb < QLEN; kb += 64) {
    __syncthreads();  // previous iteration done reading K/V tiles

    // ---- async DMA staging of K/V tiles: 2 x 64x64 bf16 = 1024 x 16B ----
    const __bf16* gk = Kten + ((size_t)bh * QLEN + kb) * DH;
    const __bf16* gv = Vten + ((size_t)bh * QLEN + kb) * DH;
    for (int i = threadIdx.x; i < 512; i += 128) {
      async_ld_b128(ldsK + i * 16, gk + i * 8);
      async_ld_b128(ldsV + i * 16, gv + i * 8);
    }
    if (threadIdx.x < 64) sm.maskb[threadIdx.x] = amask[b * QLEN + kb + threadIdx.x];
    wait_async0();
    __syncthreads();

    // ---- scores: S(16x64) = Q(16x64) . K^T, four 16-key tiles ----
#pragma unroll
    for (int t = 0; t < 4; ++t) {
      v8f s = zero8();
      s = wmma_bf16(qa0, load_bT_frag(sm.K + (t * 16) * 64 + 0, 64), s);
      s = wmma_bf16(qa1, load_bT_frag(sm.K + (t * 16) * 64 + 32, 64), s);
#pragma unroll
      for (int r = 0; r < 8; ++r)
        sm.S[w][(r + 8 * hi) * 64 + t * 16 + col] = s[r];
    }
    __syncthreads();

    // ---- online softmax over this 64-key block (2 lanes per row) ----
    {
      const int srow = lane >> 1;
      const int shalf = lane & 1;
      float mx = NEGF;
#pragma unroll
      for (int c = 0; c < 32; ++c) {
        const int cc = shalf * 32 + c;
        float sv = sm.S[w][srow * 64 + cc];
        if (sm.maskb[cc] == 0) sv = NEGF;
        mx = fmaxf(mx, sv);
      }
      mx = fmaxf(mx, __shfl_xor(mx, 1, 32));
      const float mold = sm.mrow[w][srow];
      const float mnew = fmaxf(mold, mx);
      const float alpha = __expf(mold - mnew);
      float sum = 0.0f;
#pragma unroll
      for (int c = 0; c < 32; ++c) {
        const int cc = shalf * 32 + c;
        float sv = sm.S[w][srow * 64 + cc];
        if (sm.maskb[cc] == 0) sv = NEGF;
        const float p = __expf(sv - mnew);
        sum += p;
        sm.P[w][srow * 64 + cc] = (__bf16)p;
      }
      sum += __shfl_xor(sum, 1, 32);
      if (shalf == 0) {
        sm.mrow[w][srow] = mnew;
        sm.lrow[w][srow] = sm.lrow[w][srow] * alpha + sum;
        sm.arow[w][srow] = alpha;
      }
    }
    __syncthreads();

    // ---- rescale accumulator, then context += P(16x64) x V(64x64) ----
    float al[8];
#pragma unroll
    for (int r = 0; r < 8; ++r) al[r] = sm.arow[w][r + 8 * hi];
#pragma unroll
    for (int t = 0; t < 4; ++t)
#pragma unroll
      for (int r = 0; r < 8; ++r) acc[t][r] *= al[r];

    v16bf pa0 = load_a_frag(sm.P[w], 64);
    v16bf pa1 = load_a_frag(sm.P[w] + 32, 64);
#pragma unroll
    for (int t = 0; t < 4; ++t) {
      acc[t] = wmma_bf16(pa0, load_b_frag(sm.V + 0 * 64 + t * 16, 64), acc[t]);
      acc[t] = wmma_bf16(pa1, load_b_frag(sm.V + 32 * 64 + t * 16, 64), acc[t]);
    }
  }

  __syncthreads();
  float il[8];
#pragma unroll
  for (int r = 0; r < 8; ++r) il[r] = 1.0f / sm.lrow[w][r + 8 * hi];

  // context layout: [b, q, h*64+dh]  (already the transposed/merged layout)
#pragma unroll
  for (int t = 0; t < 4; ++t)
#pragma unroll
    for (int r = 0; r < 8; ++r) {
      const int q = q0 + r + 8 * hi;
      ctx[((size_t)b * QLEN + q) * DIM + h * 64 + t * 16 + col] =
          (__bf16)(acc[t][r] * il[r]);
    }
}

// ---------------------------------------------------------------------------
// Kernel 4: output projection.  out = ctx @ Wo^T + bo   (f32 output)
// ---------------------------------------------------------------------------
__global__ void out_proj_kernel(const __bf16* __restrict__ ctx,
                                const __bf16* __restrict__ Wo,
                                const float* __restrict__ bo,
                                float* __restrict__ out) {
  const int wave = threadIdx.x >> 5;
  const int lane = threadIdx.x & 31;
  const int NT = DIM / 64;        // 16
  const int tile = blockIdx.x * 8 + wave;   // 512*16 tiles
  const int mt = tile / NT, nt = tile % NT;

  const __bf16* A0 = ctx + (size_t)mt * 16 * DIM;
  v8f acc[4] = {zero8(), zero8(), zero8(), zero8()};
  for (int kk = 0; kk < DIM; kk += 32) {
    v16bf a = load_a_frag(A0 + kk, DIM);
#pragma unroll
    for (int t = 0; t < 4; ++t) {
      v16bf bf = load_bT_frag(Wo + (size_t)(nt * 64 + t * 16) * DIM + kk, DIM);
      acc[t] = wmma_bf16(a, bf, acc[t]);
    }
  }

  const int hi = lane >> 4, col = lane & 15;
#pragma unroll
  for (int t = 0; t < 4; ++t) {
    const int n = nt * 64 + t * 16 + col;
    const float bias_n = bo[n];
#pragma unroll
    for (int r = 0; r < 8; ++r) {
      const int m = mt * 16 + r + 8 * hi;
      out[(size_t)m * DIM + n] = acc[t][r] + bias_n;
    }
  }
}

// ---------------------------------------------------------------------------
// Host launch
// ---------------------------------------------------------------------------
extern "C" void kernel_launch(void* const* d_in, const int* in_sizes, int n_in,
                              void* d_out, int out_size, void* d_ws, size_t ws_size,
                              hipStream_t stream) {
  const float* hid  = (const float*)d_in[0];
  const int*   mask = (const int*)d_in[1];
  const float* Wq   = (const float*)d_in[2];
  const float* bq   = (const float*)d_in[3];
  const float* Wk   = (const float*)d_in[4];
  const float* bk   = (const float*)d_in[5];
  const float* Wv   = (const float*)d_in[6];
  const float* bv   = (const float*)d_in[7];
  const float* Wo   = (const float*)d_in[8];
  const float* bo   = (const float*)d_in[9];
  float* out = (float*)d_out;

  const size_t nHid = (size_t)BS * QLEN * DIM;   // 8,388,608
  const size_t nW   = (size_t)DIM * DIM;         // 1,048,576

  __bf16* hidb = (__bf16*)d_ws;
  __bf16* wqb  = hidb + nHid;         // Wq,Wk,Wv,Wo stacked contiguously
  __bf16* wkb  = wqb + nW;
  __bf16* wvb  = wkb + nW;
  __bf16* wob  = wvb + nW;
  __bf16* qkvb = wob + nW;            // [3][BS*NH][QLEN][DH]
  __bf16* ctxb = qkvb + 3 * nHid;     // [BS][QLEN][DIM]   (~92 MB total ws)

  cvt_f32_bf16<<<4096, 256, 0, stream>>>(hid, hidb, (int)nHid);
  cvt_f32_bf16<<<1024, 256, 0, stream>>>(Wq, wqb, (int)nW);
  cvt_f32_bf16<<<1024, 256, 0, stream>>>(Wk, wkb, (int)nW);
  cvt_f32_bf16<<<1024, 256, 0, stream>>>(Wv, wvb, (int)nW);
  cvt_f32_bf16<<<1024, 256, 0, stream>>>(Wo, wob, (int)nW);

  // 3 mats * 512 row-tiles * 16 col-tiles = 24576 wave-tiles / 8 waves
  qkv_proj_kernel<<<3072, 256, 0, stream>>>(hidb, wqb, bq, bk, bv, qkvb);

  // BS*NH * (QLEN/64) = 2048 blocks of 4 waves
  attn_kernel<<<2048, 128, 0, stream>>>(qkvb, qkvb + nHid, qkvb + 2 * nHid,
                                        mask, ctxb);

  // 512*16 = 8192 wave-tiles / 8 waves
  out_proj_kernel<<<1024, 256, 0, stream>>>(ctxb, wob, bo, out);
}
